// VqVae_25847113187784
// MI455X (gfx1250) — matrix operand order repaired and checked by reading, not compiled, not run on hardware
//
#include <hip/hip_runtime.h>
#include <hip/hip_bf16.h>

typedef __bf16 v16bf __attribute__((ext_vector_type(16)));
typedef __bf16 v8bf  __attribute__((ext_vector_type(8)));
typedef float  v8f   __attribute__((ext_vector_type(8)));

#define NTOT   131072
#define INDIM  56
#define HDIM   512
#define DDIM   256
#define KCODE  1024
#define QNUM   2
#define ROWS   64          // rows per block
#define LDA    520         // padded bf16 LDS row stride: 1040B row -> 16B aligned + 4-bank skew
#define THREADS 128        // 4 wave32s, one 16-row strip each

// ---------------- WMMA operand loaders ----------------

// A: 16x32 bf16 fragment from LDS (row-major, stride LDA).
// lane L: row = m0+L%16, K = {h16+0..15}  ->  two ds_load_b128
static __device__ __forceinline__ v16bf loadA_lds(const __bf16* A, int m0, int k0, int lane) {
  const __bf16* base = A + (m0 + (lane & 15)) * LDA + k0 + ((lane >> 4) << 4);
  v8bf lo = *(const v8bf*)(base);
  v8bf hi = *(const v8bf*)(base + 8);
  return __builtin_shufflevector(lo, hi, 0,1,2,3,4,5,6,7,8,9,10,11,12,13,14,15);
}

// ---------------- fused GEMM layer: C = relu?(A*B + bias) -> LDS bf16 ----------------
static __device__ void gemm_layer(const __bf16* A, int Kdim,
                                  const __bf16* Bp, int ntN,
                                  const float* bias, int Ncols, bool relu,
                                  __bf16* C, int m0, int lane) {
  int hl = lane & 15;
  int h8 = (lane >> 4) << 3;
#pragma unroll 1
  for (int tn0 = 0; tn0 < (Ncols >> 4); tn0 += 8) {   // 8 N-tiles (128 cols) per chunk
    v8f acc[8] = {};
#pragma unroll 2
    for (int k0 = 0; k0 < Kdim; k0 += 32) {
      v16bf a = loadA_lds(A, m0, k0, lane);
      const __bf16* bp = Bp + (((size_t)(k0 >> 5) * ntN + tn0) << 9) + (lane << 4);
#pragma unroll
      for (int j = 0; j < 8; ++j) {
        v16bf b = *(const v16bf*)(bp + (j << 9));
        acc[j] = __builtin_amdgcn_wmma_f32_16x16x32_bf16(
                     false, a, false, b, (short)0, acc[j], false, false);
      }
    }
#pragma unroll
    for (int j = 0; j < 8; ++j) {
      int n = ((tn0 + j) << 4) + hl;
      float bv = bias[n];
#pragma unroll
      for (int r = 0; r < 8; ++r) {
        float vv = acc[j][r] + bv;
        if (relu) vv = fmaxf(vv, 0.f);
        C[(m0 + h8 + r) * LDA + n] = (__bf16)vv;
      }
    }
  }
}

// ---------------- VQ argmin over 1024 codes: min_n (||c_n||^2 - 2 r.c_n) ----------------
static __device__ void vq_argmin(const __bf16* res, const __bf16* cbp, const float* cnorm,
                                 int m0, int lane, int* idxbuf) {
  float minv[8]; int mini[8];
#pragma unroll
  for (int r = 0; r < 8; ++r) { minv[r] = 3.4e38f; mini[r] = 0; }
#pragma unroll 1
  for (int tn0 = 0; tn0 < (KCODE >> 4); tn0 += 8) {
    v8f acc[8] = {};
#pragma unroll 2
    for (int k0 = 0; k0 < DDIM; k0 += 32) {
      v16bf a = loadA_lds(res, m0, k0, lane);
      const __bf16* bp = cbp + (((size_t)(k0 >> 5) * (KCODE >> 4) + tn0) << 9) + (lane << 4);
#pragma unroll
      for (int j = 0; j < 8; ++j) {
        v16bf b = *(const v16bf*)(bp + (j << 9));
        acc[j] = __builtin_amdgcn_wmma_f32_16x16x32_bf16(
                     false, a, false, b, (short)0, acc[j], false, false);
      }
    }
#pragma unroll
    for (int j = 0; j < 8; ++j) {
      int n = ((tn0 + j) << 4) + (lane & 15);
      float cn = cnorm[n];
#pragma unroll
      for (int r = 0; r < 8; ++r) {
        float v = cn - 2.0f * acc[j][r];
        if (v < minv[r]) { minv[r] = v; mini[r] = n; }
      }
    }
  }
  // reduce across the 16 lanes that hold the same 8 rows (xor<16 stays within half)
#pragma unroll
  for (int off = 1; off < 16; off <<= 1) {
#pragma unroll
    for (int r = 0; r < 8; ++r) {
      float ov = __shfl_xor(minv[r], off, 32);
      int   oi = __shfl_xor(mini[r], off, 32);
      if (ov < minv[r] || (ov == minv[r] && oi < mini[r])) { minv[r] = ov; mini[r] = oi; }
    }
  }
  int h8 = (lane >> 4) << 3;
  if ((lane & 15) == 0) {
#pragma unroll
    for (int r = 0; r < 8; ++r) idxbuf[m0 + h8 + r] = mini[r];
  }
}

// ---------------- fused pipeline kernel ----------------
__global__ void __launch_bounds__(THREADS)
vqvae_fused(const float* __restrict__ x,
            const float* __restrict__ eb1, const float* __restrict__ eb2,
            const float* __restrict__ eb3, const float* __restrict__ db1,
            const float* __restrict__ db2, const float* __restrict__ db3,
            const float* __restrict__ cbs, const float* __restrict__ cnorm,
            const __bf16* __restrict__ ew1p, const __bf16* __restrict__ ew2p,
            const __bf16* __restrict__ ew3p, const __bf16* __restrict__ dw1p,
            const __bf16* __restrict__ dw2p, const __bf16* __restrict__ dw3p,
            const __bf16* __restrict__ cbp,
            float* __restrict__ out, float* __restrict__ acc) {
  extern __shared__ __bf16 smem[];
  __bf16* bufA = smem;
  __bf16* bufF = smem + ROWS * LDA;

  int lane  = threadIdx.x & 31;
  int wave  = threadIdx.x >> 5;
  int m0    = wave * 16;
  int gbase = blockIdx.x * ROWS;

  // ---- stage x (f32 -> bf16) into bufF cols [0,64), zero-padded 56->64 ----
#pragma unroll 1
  for (int row = 0; row < 16; ++row) {
    int grow = m0 + row;
#pragma unroll
    for (int j = 0; j < 2; ++j) {
      int col = lane + 32*j;
      float v = (col < INDIM) ? x[(size_t)(gbase + grow) * INDIM + col] : 0.f;
      bufF[grow * LDA + col] = (__bf16)v;
    }
  }

  // ---- encoder MLP ----
  gemm_layer(bufF,  64, ew1p, HDIM >> 4, eb1, HDIM, true,  bufA, m0, lane);
  gemm_layer(bufA, HDIM, ew2p, HDIM >> 4, eb2, HDIM, true,  bufF, m0, lane);
  gemm_layer(bufF, HDIM, ew3p, DDIM >> 4, eb3, DDIM, false, bufA, m0, lane);
  // bufA cols[0,256): z (residual); cols[256,512): quantized accumulator

  // ---- residual VQ (Q=2) ----
  int* idxbuf = (int*)bufF;            // bufF dead until decoder L1
  float vq_partial = 0.f;
  int c0 = lane << 3;                  // this lane's contiguous 8-column slice
#pragma unroll 1
  for (int iq = 0; iq < QNUM; ++iq) {
    const float*  cb  = cbs + (size_t)iq * KCODE * DDIM;
    const __bf16* cbq = cbp + (size_t)iq * KCODE * DDIM;
    vq_argmin(bufA, cbq, cnorm + iq * KCODE, m0, lane, idxbuf);
#pragma unroll 1
    for (int row = 0; row < 16; ++row) {
      int grow = m0 + row;
      int code = idxbuf[grow];
      const float* q = cb + (size_t)code * DDIM + c0;
      float4 q01 = *(const float4*)(q);
      float4 q23 = *(const float4*)(q + 4);
      float qv[8] = {q01.x, q01.y, q01.z, q01.w, q23.x, q23.y, q23.z, q23.w};
      __bf16* resp = bufA + grow * LDA + c0;
#pragma unroll
      for (int j = 0; j < 8; ++j) {
        float rv = (float)resp[j];
        float d  = qv[j] - rv;
        vq_partial += d * d;                             // commitment partial
        if (iq == 0) {
          resp[j]       = (__bf16)(rv - qv[j]);          // residual
          resp[256 + j] = (__bf16)qv[j];                 // quantized init
        } else {
          float prev = (float)resp[256 + j];
          resp[256 + j] = (__bf16)(prev + qv[j]);        // quantized accum
        }
      }
      if (lane == 0)
        out[1 + (size_t)(gbase + grow) * QNUM + iq] = (float)code;
    }
  }

  // ---- decoder MLP ----
  gemm_layer(bufA + 256, DDIM, dw1p, HDIM >> 4, db1, HDIM, true, bufF, m0, lane);
  gemm_layer(bufF,       HDIM, dw2p, HDIM >> 4, db2, HDIM, true, bufA, m0, lane);

  // ---- decoder L3 (Ncols padded 56->64) + L1 reconstruction loss ----
  float l1 = 0.f;
  {
    v8f a4[4] = {};
#pragma unroll 2
    for (int k0 = 0; k0 < HDIM; k0 += 32) {
      v16bf a = loadA_lds(bufA, m0, k0, lane);
      const __bf16* bp = dw3p + (((size_t)(k0 >> 5) * 4) << 9) + (lane << 4);
#pragma unroll
      for (int j = 0; j < 4; ++j) {
        v16bf b = *(const v16bf*)(bp + (j << 9));
        a4[j] = __builtin_amdgcn_wmma_f32_16x16x32_bf16(
                    false, a, false, b, (short)0, a4[j], false, false);
      }
    }
    int h8 = (lane >> 4) << 3;
#pragma unroll
    for (int j = 0; j < 4; ++j) {
      int n = 16*j + (lane & 15);
      if (n < INDIM) {
        float bv = db3[n];
#pragma unroll
        for (int r = 0; r < 8; ++r) {
          size_t grow = (size_t)gbase + m0 + h8 + r;
          float vv = a4[j][r] + bv;
          l1 += fabsf(x[grow * INDIM + n] - vv);
        }
      }
    }
  }

  // ---- wave reduce + global accumulate ----
  float s0 = vq_partial, s1 = l1;
#pragma unroll
  for (int off = 16; off; off >>= 1) {
    s0 += __shfl_xor(s0, off, 32);
    s1 += __shfl_xor(s1, off, 32);
  }
  if (lane == 0) { atomicAdd(&acc[0], s0); atomicAdd(&acc[1], s1); }
}

// ---------------- prep kernels (run once per launch) ----------------

// Pack f32 matrix (K x N, or transposed source) into WMMA-fragment-major bf16,
// zero-padding K -> Kp and N -> Np. One thread per (tile, lane).
__global__ void pack_b(const float* __restrict__ src, int lds, int Kvalid, int Nvalid,
                       int Kp, int Np, int transpose, __bf16* __restrict__ dst) {
  int t    = blockIdx.x * blockDim.x + threadIdx.x;
  int lane = t & 31;
  int tile = t >> 5;
  int ntN = Np >> 4;
  int ntK = Kp >> 5;
  if (tile >= ntK * ntN) return;
  int tk = tile / ntN, tn = tile - tk * ntN;
  int n  = (tn << 4) + (lane & 15);
  int kh = (lane >> 4) << 4;
  __bf16 vals[16];
#pragma unroll
  for (int v = 0; v < 8; ++v) {
    int k = (tk << 5) + kh + 2*v;
    float f0 = 0.f, f1 = 0.f;
    if (n < Nvalid) {
      if (k     < Kvalid) f0 = transpose ? src[(size_t)n*lds + k]     : src[(size_t)k*lds + n];
      if (k + 1 < Kvalid) f1 = transpose ? src[(size_t)n*lds + k + 1] : src[(size_t)(k+1)*lds + n];
    }
    vals[2*v] = (__bf16)f0; vals[2*v+1] = (__bf16)f1;
  }
  __bf16* o = dst + ((size_t)tile << 9) + (lane << 4);
#pragma unroll
  for (int i = 0; i < 16; ++i) o[i] = vals[i];
}

__global__ void vqvae_init(float* acc) {
  if (threadIdx.x < 2) acc[threadIdx.x] = 0.f;
}

__global__ void vqvae_cnorm(const float* __restrict__ cbs, float* __restrict__ cnorm) {
  int i = blockIdx.x * blockDim.x + threadIdx.x;
  if (i < QNUM * KCODE) {
    const float* r = cbs + (size_t)i * DDIM;
    float s = 0.f;
#pragma unroll 4
    for (int j = 0; j < DDIM; j += 4) {
      float4 v = *(const float4*)(r + j);
      s += v.x*v.x + v.y*v.y + v.z*v.z + v.w*v.w;
    }
    cnorm[i] = s;
  }
}

__global__ void vqvae_final(const float* acc, float* out) {
  if (threadIdx.x == 0) {
    float vq = acc[0] / (float)((size_t)NTOT * DDIM);
    float l1 = acc[1] / (float)((size_t)NTOT * INDIM);
    out[0] = l1 + 5.0f * vq;
  }
}

// ---------------- launcher ----------------
extern "C" void kernel_launch(void* const* d_in, const int* in_sizes, int n_in,
                              void* d_out, int out_size, void* d_ws, size_t ws_size,
                              hipStream_t stream) {
  const float* x   = (const float*)d_in[0];
  const float* ew1 = (const float*)d_in[1];
  const float* eb1 = (const float*)d_in[2];
  const float* ew2 = (const float*)d_in[3];
  const float* eb2 = (const float*)d_in[4];
  const float* ew3 = (const float*)d_in[5];
  const float* eb3 = (const float*)d_in[6];
  const float* dw1 = (const float*)d_in[7];
  const float* db1 = (const float*)d_in[8];
  const float* dw2 = (const float*)d_in[9];
  const float* db2 = (const float*)d_in[10];
  const float* dw3 = (const float*)d_in[11];
  const float* db3 = (const float*)d_in[12];
  const float* cbs = (const float*)d_in[13];
  float* out = (float*)d_out;

  char* ws = (char*)d_ws;
  float*  acc   = (float*)(ws);            // 2 floats (256B reserved)
  float*  cnorm = (float*)(ws + 256);      // Q*K floats (8192B)
  __bf16* ew1p  = (__bf16*)(ws + 8448);    // packed fragment-major weights:
  __bf16* ew2p  = ew1p + 64  * 512;        //  64x512
  __bf16* ew3p  = ew2p + 512 * 512;        // 512x512
  __bf16* dw1p  = ew3p + 512 * 256;        // 512x256
  __bf16* dw2p  = dw1p + 256 * 512;        // 256x512
  __bf16* dw3p  = dw2p + 512 * 512;        // 512x512
  __bf16* cbp   = dw3p + 512 * 64;         // 512x64 ; then 2 x (256x1024) codebooks^T

  vqvae_init<<<1, 64, 0, stream>>>(acc);
  vqvae_cnorm<<<(QNUM * KCODE + 255) / 256, 256, 0, stream>>>(cbs, cnorm);

  auto np = [&](int Kp, int Np) { return ((Kp >> 5) * (Np >> 4) * 32 + 255) / 256; };
  pack_b<<<np( 64, 512), 256, 0, stream>>>(ew1, 512,  56,  512,  64,  512, 0, ew1p);
  pack_b<<<np(512, 512), 256, 0, stream>>>(ew2, 512, 512,  512, 512,  512, 0, ew2p);
  pack_b<<<np(512, 256), 256, 0, stream>>>(ew3, 256, 512,  256, 512,  256, 0, ew3p);
  pack_b<<<np(256, 512), 256, 0, stream>>>(dw1, 512, 256,  512, 256,  512, 0, dw1p);
  pack_b<<<np(512, 512), 256, 0, stream>>>(dw2, 512, 512,  512, 512,  512, 0, dw2p);
  pack_b<<<np(512,  64), 256, 0, stream>>>(dw3,  56, 512,   56, 512,   64, 0, dw3p);
  pack_b<<<np(256,1024), 256, 0, stream>>>(cbs,               256, 256, 1024, 256, 1024, 1, cbp);
  pack_b<<<np(256,1024), 256, 0, stream>>>(cbs + 1024 * 256,  256, 256, 1024, 256, 1024, 1,
                                           cbp + 256 * 1024);

  size_t smem = (size_t)2 * ROWS * LDA * sizeof(__bf16);   // 133120 B dynamic LDS
  vqvae_fused<<<NTOT / ROWS, THREADS, smem, stream>>>(
      x, eb1, eb2, eb3, db1, db2, db3, cbs, cnorm,
      ew1p, ew2p, ew3p, dw1p, dw2p, dw3p, cbp, out, acc);

  vqvae_final<<<1, 32, 0, stream>>>(acc, out);
}
